// SelfAttention_18013092839477
// MI455X (gfx1250) — compile-verified, matrix operand order checked by reading, not run on hardware
//
#include <hip/hip_runtime.h>
#include <hip/hip_bf16.h>

// ---------------------------------------------------------------------------
// Self-attention for B=4, S=2048, D=1024 (fp32 in/out) on gfx1250.
// bf16 WMMA for all three matmul stages; fused scores+softmax+PV keeps the
// 4x2048x2048 score tensor in LDS.  Round 4: x-tile staging in the projection
// GEMM exercises BOTH CDNA5 DMA paths: TDM (tensor_load_to_lds, TENSORcnt,
// 6-arg builtin form) for rows 0-7 and global_load_async_to_lds_b128
// (ASYNCcnt) for rows 8-15.
// ---------------------------------------------------------------------------

typedef __attribute__((ext_vector_type(16))) __bf16 v16bf;
typedef __attribute__((ext_vector_type(8)))  float  v8f;
typedef __attribute__((ext_vector_type(4)))  unsigned int u32x4;
typedef __attribute__((ext_vector_type(8)))  int i32x8;
typedef __attribute__((ext_vector_type(4)))  int i32x4;

#define D_DIM   1024
#define S_DIM   2048
#define B_DIM   4
#define ROWS    (B_DIM * S_DIM)        // 8192

__device__ __forceinline__ v8f wmma_bf16(v16bf a, v16bf b, v8f c) {
  return __builtin_amdgcn_wmma_f32_16x16x32_bf16(
      false, a, false, b, (short)0, c, false, false);
}

// ---------------------------------------------------------------------------
// Kernel 1a: fp32 -> bf16 elementwise convert (for x)
// ---------------------------------------------------------------------------
__global__ void cvt_f32_bf16_kernel(const float* __restrict__ in,
                                    __bf16* __restrict__ out, int n) {
  int i = blockIdx.x * blockDim.x + threadIdx.x;
  int stride = gridDim.x * blockDim.x;
  for (; i < n; i += stride) out[i] = (__bf16)in[i];
}

// ---------------------------------------------------------------------------
// Kernel 1b: fp32 (d,f) -> bf16 transposed (f,d) for the weight matrices.
// ---------------------------------------------------------------------------
__global__ void transpose_bf16_kernel(const float* __restrict__ W,
                                      __bf16* __restrict__ Wt) {
  int i = blockIdx.x * blockDim.x + threadIdx.x;
  int stride = gridDim.x * blockDim.x;
  for (; i < D_DIM * D_DIM; i += stride) {
    int d = i >> 10;
    int f = i & (D_DIM - 1);
    Wt[(f << 10) + d] = (__bf16)W[i];
  }
}

// ---------------------------------------------------------------------------
// Kernel 2: QKV projection GEMM.  Y = x * W + b  (8192x1024)x(1024x1024).
// grid.x = row tiles (512), grid.y = which matrix (0=Q, 1=K, 2=V).
// The 16x1024 x-tile (32 KB) is shared by all 8 waves: rows 0-7 staged by a
// single TDM descriptor (wave 0), rows 8-15 by per-lane async loads.  Each
// wave then computes a 16x64 strip (4 accumulators per A-fragment).
// ---------------------------------------------------------------------------
__global__ __launch_bounds__(256)
void qkv_gemm_kernel(const __bf16* __restrict__ xb,    // [8192][1024]
                     const __bf16* __restrict__ Wt,    // [3][1024][1024] (f,d)
                     const float* __restrict__ bq,
                     const float* __restrict__ bk,
                     const float* __restrict__ bvec,
                     __bf16* __restrict__ Qb,          // [8192][1024]
                     __bf16* __restrict__ Kb,          // [8192][1024]
                     __bf16* __restrict__ Vt) {        // [4][1024][2048]
  __shared__ __attribute__((aligned(128))) __bf16 sX[16 * D_DIM];  // 32 KB

  const int m0    = blockIdx.x << 4;
  const int which = blockIdx.y;
  const int wave  = threadIdx.x >> 5;
  const int lane  = threadIdx.x & 31;
  const int rowm  = lane & 15;
  const int khalf = (lane >> 4) << 4;

  // ---- stage the x tile into LDS ------------------------------------------
  {
    const char* gbase = (const char*)(xb + (size_t)m0 * D_DIM);
    const uint32_t lds_base = (uint32_t)(uintptr_t)(&sX[0]);

    if (wave == 0) {
      // TDM: rows 0..7 = 8 x 2048 B tile via one Tensor-DMA descriptor.
      // D# group0: count=1 | lds_addr | global_addr | type=2
      const uint64_t ga = (uint64_t)(uintptr_t)gbase;
      u32x4 g0;
      g0.x = 1u;                                            // count = 1
      g0.y = lds_base;                                      // LDS dest (bytes)
      g0.z = (uint32_t)ga;                                  // global addr lo
      g0.w = (uint32_t)((ga >> 32) & 0x01FFFFFFull) | 0x80000000u;  // type=2
      // D# group1: data_size=8B; tensor_dim0=256 units; tensor_dim1=8192;
      // tile_dim0=256; tile_dim1=8; tensor_dim0_stride=256 units (2048 B)
      i32x8 g1;
      g1[0] = 0x00030000;   // workgroup_mask=0, data_size=3 (8 B)
      g1[1] = 0x01000000;   // tensor_dim0[15:0]=256 in bits[31:16]
      g1[2] = 0x20000000;   // tensor_dim0 hi=0; tensor_dim1[15:0]=8192
      g1[3] = 0x01000000;   // tensor_dim1 hi=0; tile_dim0=256 in bits[31:16]
      g1[4] = 8;            // tile_dim1 = 8 rows; tile_dim2 = 0
      g1[5] = 256;          // tensor_dim0_stride lo = 256 units
      g1[6] = 0;            // stride0 hi, tensor_dim1_stride lo
      g1[7] = 0;
      i32x4 gz4 = {0, 0, 0, 0};
      i32x8 gz8 = {0, 0, 0, 0, 0, 0, 0, 0};
      __builtin_amdgcn_tensor_load_to_lds(g0, g1, gz4, gz4, gz8, 0);
      __builtin_amdgcn_s_wait_tensorcnt(0);
    }

    // Async path: rows 8..15 (16 KB) = 1024 x 16B chunks, 4 per thread.
    const char* gbase2 = gbase + 8 * 2048;
#pragma unroll
    for (int j = 0; j < 4; ++j) {
      const int c = ((int)threadIdx.x + (j << 8)) << 4;     // byte offset
      const uint64_t ga = (uint64_t)(uintptr_t)(gbase2 + c);
      const uint32_t la = lds_base + 16384u + (uint32_t)c;
      asm volatile("global_load_async_to_lds_b128 %0, %1, off"
                   :: "v"(la), "v"(ga) : "memory");
    }
    asm volatile("s_wait_asynccnt 0x0" ::: "memory");
  }
  __syncthreads();

  const __bf16* ArowL = sX + (size_t)rowm * D_DIM + khalf;   // LDS A-fragments
  const __bf16* Wsel  = Wt + (size_t)which * (D_DIM * D_DIM);
  const float*  bias  = (which == 0) ? bq : (which == 1) ? bk : bvec;
  const int     mbase = (lane < 16) ? 0 : 8;

  for (int g = 0; g < 2; ++g) {               // 2 strips of 4 n-tiles per wave
    const int nt0 = (wave << 3) + (g << 2);
    const __bf16* Bbase =
        Wsel + (size_t)((nt0 << 4) + rowm) * D_DIM + khalf;
    v8f acc0 = {}, acc1 = {}, acc2 = {}, acc3 = {};
#pragma unroll 2
    for (int f0 = 0; f0 < D_DIM; f0 += 32) {
      __builtin_prefetch(Bbase + f0 + 256);
      v16bf a  = *(const v16bf*)(ArowL + f0);               // ds_load
      v16bf b0 = *(const v16bf*)(Bbase + f0);
      v16bf b1 = *(const v16bf*)(Bbase + 16 * D_DIM + f0);
      v16bf b2 = *(const v16bf*)(Bbase + 32 * D_DIM + f0);
      v16bf b3 = *(const v16bf*)(Bbase + 48 * D_DIM + f0);
      acc0 = wmma_bf16(a, b0, acc0);
      acc1 = wmma_bf16(a, b1, acc1);
      acc2 = wmma_bf16(a, b2, acc2);
      acc3 = wmma_bf16(a, b3, acc3);
    }
    v8f accs[4] = {acc0, acc1, acc2, acc3};
    if (which == 2) {                         // V: store transposed (f,t)
#pragma unroll
      for (int q = 0; q < 4; ++q) {
        const int n = ((nt0 + q) << 4) + rowm;
        const float bb = bias[n];
#pragma unroll
        for (int i = 0; i < 8; ++i) {
          const int r = m0 + mbase + i;
          const int bidx = r >> 11;
          const int t = r & (S_DIM - 1);
          Vt[((size_t)bidx << 21) + ((size_t)n << 11) + t] =
              (__bf16)(accs[q][i] + bb);
        }
      }
    } else {
      __bf16* Y = (which == 0) ? Qb : Kb;
#pragma unroll
      for (int q = 0; q < 4; ++q) {
        const int n = ((nt0 + q) << 4) + rowm;
        const float bb = bias[n];
#pragma unroll
        for (int i = 0; i < 8; ++i)
          Y[(size_t)(m0 + mbase + i) * D_DIM + n] = (__bf16)(accs[q][i] + bb);
      }
    }
  }
}

// ---------------------------------------------------------------------------
// Kernel 3: fused attention.  One workgroup = (batch, 16-query-row tile).
//   Phase 1: S = Q K^T / 32  -> bf16 strip [16][2048] in LDS (exactly 64 KB),
//            4 key-tiles per wave iteration (one A-fragment -> 4 WMMAs)
//   Phase 2: softmax in place (shfl_xor over 16-lane groups)
//   Phase 3: O = P V, A-fragments from LDS, 4 feature-tiles per iteration
// ---------------------------------------------------------------------------
__global__ __launch_bounds__(256)
void attn_kernel(const __bf16* __restrict__ Qb,   // [4][2048][1024]
                 const __bf16* __restrict__ Kb,   // [4][2048][1024]
                 const __bf16* __restrict__ Vt,   // [4][1024][2048]
                 float* __restrict__ out) {       // [4][2048][1024] f32
  __shared__ __attribute__((aligned(128))) __bf16 sP[16 * S_DIM];  // 64 KB

  const int tile  = blockIdx.x;
  const int bidx  = blockIdx.y;
  const int wave  = threadIdx.x >> 5;
  const int lane  = threadIdx.x & 31;
  const int rowm  = lane & 15;
  const int khalf = (lane >> 4) << 4;
  const int m0    = tile << 4;
  const int mbase = (lane < 16) ? 0 : 8;

  const __bf16* Qbase =
      Qb + ((size_t)bidx * S_DIM + m0 + rowm) * D_DIM + khalf;
  const __bf16* Kbase = Kb + (size_t)bidx * S_DIM * D_DIM;

  // ---- Phase 1: score strip, 4 key tiles per group -------------------------
  for (int g = 0; g < 4; ++g) {               // 16 tiles per wave, 4 groups
    const int tt0 = (wave << 4) + (g << 2);
    const __bf16* Bbase =
        Kbase + (size_t)((tt0 << 4) + rowm) * D_DIM + khalf;
    v8f acc0 = {}, acc1 = {}, acc2 = {}, acc3 = {};
#pragma unroll 2
    for (int f0 = 0; f0 < D_DIM; f0 += 32) {
      __builtin_prefetch(Bbase + f0 + 256);
      v16bf a  = *(const v16bf*)(Qbase + f0);
      v16bf b0 = *(const v16bf*)(Bbase + f0);
      v16bf b1 = *(const v16bf*)(Bbase + 16 * D_DIM + f0);
      v16bf b2 = *(const v16bf*)(Bbase + 32 * D_DIM + f0);
      v16bf b3 = *(const v16bf*)(Bbase + 48 * D_DIM + f0);
      acc0 = wmma_bf16(a, b0, acc0);
      acc1 = wmma_bf16(a, b1, acc1);
      acc2 = wmma_bf16(a, b2, acc2);
      acc3 = wmma_bf16(a, b3, acc3);
    }
    v8f accs[4] = {acc0, acc1, acc2, acc3};
#pragma unroll
    for (int q = 0; q < 4; ++q) {
      const int n = ((tt0 + q) << 4) + rowm;
#pragma unroll
      for (int i = 0; i < 8; ++i)
        sP[(mbase + i) * S_DIM + n] = (__bf16)(accs[q][i] * 0.03125f);
    }
  }
  __syncthreads();

  // ---- Phase 2: softmax, 16 threads per row --------------------------------
  {
    const int row = threadIdx.x >> 4;
    const int c0  = threadIdx.x & 15;
    __bf16* rp = sP + row * S_DIM;

    float mx = -3.0e38f;
    for (int c = c0; c < S_DIM; c += 16) mx = fmaxf(mx, (float)rp[c]);
    for (int off = 8; off; off >>= 1) mx = fmaxf(mx, __shfl_xor(mx, off, 16));

    float sum = 0.0f;
    for (int c = c0; c < S_DIM; c += 16) {
      float e = __expf((float)rp[c] - mx);
      sum += e;
      rp[c] = (__bf16)e;
    }
    for (int off = 8; off; off >>= 1) sum += __shfl_xor(sum, off, 16);

    const float inv = 1.0f / sum;
    for (int c = c0; c < S_DIM; c += 16)
      rp[c] = (__bf16)((float)rp[c] * inv);
  }
  __syncthreads();

  // ---- Phase 3: O = P * V, 4 feature tiles per group -----------------------
  const __bf16* VTb = Vt + ((size_t)bidx << 21);          // [1024][2048]
  const __bf16* Pl  = sP + (size_t)rowm * S_DIM + khalf;  // LDS A-fragments
  for (int g = 0; g < 2; ++g) {               // 8 tiles per wave, 2 groups
    const int nt0 = (wave << 3) + (g << 2);
    const __bf16* Bbase =
        VTb + (size_t)((nt0 << 4) + rowm) * S_DIM + khalf;
    v8f acc0 = {}, acc1 = {}, acc2 = {}, acc3 = {};
#pragma unroll 2
    for (int t0 = 0; t0 < S_DIM; t0 += 32) {
      __builtin_prefetch(Bbase + t0 + 256);
      v16bf a  = *(const v16bf*)(Pl + t0);                // ds_load
      v16bf b0 = *(const v16bf*)(Bbase + t0);
      v16bf b1 = *(const v16bf*)(Bbase + 16 * S_DIM + t0);
      v16bf b2 = *(const v16bf*)(Bbase + 32 * S_DIM + t0);
      v16bf b3 = *(const v16bf*)(Bbase + 48 * S_DIM + t0);
      acc0 = wmma_bf16(a, b0, acc0);
      acc1 = wmma_bf16(a, b1, acc1);
      acc2 = wmma_bf16(a, b2, acc2);
      acc3 = wmma_bf16(a, b3, acc3);
    }
    v8f accs[4] = {acc0, acc1, acc2, acc3};
#pragma unroll
    for (int q = 0; q < 4; ++q) {
      const int n = ((nt0 + q) << 4) + rowm;
#pragma unroll
      for (int i = 0; i < 8; ++i)
        out[((size_t)bidx * S_DIM + m0 + mbase + i) * D_DIM + n] = accs[q][i];
    }
  }
}

// ---------------------------------------------------------------------------
// Launch wrapper.  Workspace (bf16): xb@0 (16MB) | Wt@16MB (6MB) | Qb@22MB
// (16MB) | Kb@38MB (16MB) | Vt@54MB (16MB).  Total 70 MB.
// ---------------------------------------------------------------------------
extern "C" void kernel_launch(void* const* d_in, const int* in_sizes, int n_in,
                              void* d_out, int out_size, void* d_ws,
                              size_t ws_size, hipStream_t stream) {
  const float* x  = (const float*)d_in[0];
  const float* Wq = (const float*)d_in[1];
  const float* bq = (const float*)d_in[2];
  const float* Wk = (const float*)d_in[3];
  const float* bk = (const float*)d_in[4];
  const float* Wv = (const float*)d_in[5];
  const float* bv = (const float*)d_in[6];
  float* out = (float*)d_out;

  char* ws = (char*)d_ws;
  __bf16* xb = (__bf16*)(ws);
  __bf16* Wt = (__bf16*)(ws + (16ull << 20));
  __bf16* Qb = (__bf16*)(ws + (22ull << 20));
  __bf16* Kb = (__bf16*)(ws + (38ull << 20));
  __bf16* Vt = (__bf16*)(ws + (54ull << 20));

  cvt_f32_bf16_kernel<<<8192, 256, 0, stream>>>(x, xb, ROWS * D_DIM);
  transpose_bf16_kernel<<<1024, 256, 0, stream>>>(Wq, Wt + 0ull * D_DIM * D_DIM);
  transpose_bf16_kernel<<<1024, 256, 0, stream>>>(Wk, Wt + 1ull * D_DIM * D_DIM);
  transpose_bf16_kernel<<<1024, 256, 0, stream>>>(Wv, Wt + 2ull * D_DIM * D_DIM);

  dim3 g2(ROWS / 16, 3);
  qkv_gemm_kernel<<<g2, 256, 0, stream>>>(xb, Wt, bq, bk, bv, Qb, Kb, Vt);

  dim3 g3(S_DIM / 16, B_DIM);
  attn_kernel<<<g3, 256, 0, stream>>>(Qb, Kb, Vt, out);
}
